// SparseConvNet_31963146616882
// MI455X (gfx1250) — compile-verified, matrix operand order checked by reading, not run on hardware
//
#include <hip/hip_runtime.h>
#include <hip/hip_bf16.h>

typedef __attribute__((ext_vector_type(16))) _Float16 v16h;
typedef __attribute__((ext_vector_type(8)))  _Float16 v8h;
typedef __attribute__((ext_vector_type(8)))  float    v8f;
typedef __attribute__((ext_vector_type(4)))  float    v4f;

__device__ __forceinline__ v16h combine8(v8h lo, v8h hi) {
  v16h r;
#pragma unroll
  for (int i = 0; i < 8; ++i) { r[i] = lo[i]; r[8 + i] = hi[i]; }
  return r;
}

// ---------------------------------------------------------------------------
// mask -> float
__global__ void k_mask_to_f32(const int* __restrict__ m, float* __restrict__ mf, int n) {
  int i = blockIdx.x * blockDim.x + threadIdx.x;
  if (i < n) mf[i] = (float)m[i];
}

// 3x3x3 max-pool, stride 2, pad 1 : 128^3 -> 64^3
__global__ void k_mask_down(const float* __restrict__ m0, float* __restrict__ m1) {
  int i = blockIdx.x * blockDim.x + threadIdx.x;
  if (i >= 64 * 64 * 64) return;
  int x = i & 63, y = (i >> 6) & 63, z = i >> 12;
  float r = 0.f;
  for (int dz = -1; dz <= 1; ++dz) {
    int zi = 2 * z + dz; if (zi < 0 || zi > 127) continue;
    for (int dy = -1; dy <= 1; ++dy) {
      int yi = 2 * y + dy; if (yi < 0 || yi > 127) continue;
      for (int dx = -1; dx <= 1; ++dx) {
        int xi = 2 * x + dx; if (xi < 0 || xi > 127) continue;
        r = fmaxf(r, m0[(zi * 128 + yi) * 128 + xi]);
      }
    }
  }
  m1[i] = r;
}

// ---------------------------------------------------------------------------
// Weight repack: (COUT, CIN, 3,3,3) f32 -> fp16 [n][KP], k = tap*CIN + ci,
// zero-padded to KP (multiple of 32).  Contiguous K chunks for b128 B-loads.
__global__ void k_pack_w(const float* __restrict__ w, _Float16* __restrict__ wp,
                         int CIN, int COUT, int KP) {
  int i = blockIdx.x * blockDim.x + threadIdx.x;
  if (i >= COUT * KP) return;
  int n = i / KP, k = i % KP;
  int K = CIN * 27;
  _Float16 v = (_Float16)0.f;
  if (k < K) {
    int t = k / CIN, ci = k % CIN;
    v = (_Float16)w[(n * CIN + ci) * 27 + t];
  }
  wp[i] = v;
}

// ---------------------------------------------------------------------------
// conv0a: 3 -> 16 channels, direct f32 (K = 81, cheap), input masked at source,
// fused BN+ReLU+mask epilogue.  Output NHWC fp16: [z][y][x][16] -> two b128
// stores per voxel, lane-contiguous.
__global__ void __launch_bounds__(256)
k_conv0a(const float* __restrict__ x, const float* __restrict__ m0,
         const float* __restrict__ w, const float* __restrict__ sc,
         const float* __restrict__ sh, _Float16* __restrict__ out) {
  const int D = 128;
  __shared__ float sW[16 * 81];
  for (int i = threadIdx.x; i < 16 * 81; i += blockDim.x) sW[i] = w[i];
  __syncthreads();
  int vox = blockIdx.x * blockDim.x + threadIdx.x;
  int xx = vox & 127, yy = (vox >> 7) & 127, zz = vox >> 14;
  float acc[16];
#pragma unroll
  for (int c = 0; c < 16; ++c) acc[c] = 0.f;
  for (int ci = 0; ci < 3; ++ci) {
    for (int t = 0; t < 27; ++t) {
      int dz = t / 9, r9 = t % 9, dy = r9 / 3, dx = r9 % 3;
      int zi = zz + dz - 1, yi = yy + dy - 1, xi = xx + dx - 1;
      float v = 0.f;
      if (zi >= 0 && zi < D && yi >= 0 && yi < D && xi >= 0 && xi < D) {
        int s = (zi * D + yi) * D + xi;
        v = x[ci * D * D * D + s] * m0[s];
      }
#pragma unroll
      for (int c = 0; c < 16; ++c) acc[c] += v * sW[c * 81 + ci * 27 + t];
    }
  }
  float mv = m0[(zz * D + yy) * D + xx];
  v8h lo, hi;
#pragma unroll
  for (int c = 0; c < 8; ++c) {
    lo[c] = (_Float16)(fmaxf(acc[c] * sc[c] + sh[c], 0.f) * mv);
    hi[c] = (_Float16)(fmaxf(acc[c + 8] * sc[c + 8] + sh[c + 8], 0.f) * mv);
  }
  v8h* o = (v8h*)&out[(size_t)((zz * D + yy) * D + xx) * 16];
  o[0] = lo;
  o[1] = hi;
}

// ---------------------------------------------------------------------------
// Implicit-GEMM 3x3x3 conv, WMMA f16->f32, NHWC feature maps.
//   K ordering: k = tap*CIN + ci (channel-minor) -> contiguous 32-K chunks.
//   A: LDS halo tile (channel innermost, identical layout to NHWC global) ->
//      tile load is pure global_load_b128 -> ds_store_b128; per chunk each
//      lane issues two ds_load_b128 (runs hl*8 and 16+hl*8, per 16-bit A
//      fragment layout).
//   B: pre-packed fp16 [n][KP]; two global_load_b128 per chunk per N-tile.
//   MT = M tiles per wave (B reuse); NT = COUT/16 N tiles.
//   Epilogue fuses BN + ReLU + mask; NHWC stores are lane-contiguous.
template <int CIN, int COUT, int STRIDE, int MT, typename OUT_T>
__global__ void __launch_bounds__(128)
k_conv_wmma(const _Float16* __restrict__ in, const _Float16* __restrict__ wPack,
            const float* __restrict__ sc, const float* __restrict__ sh,
            const float* __restrict__ maskOut, OUT_T* __restrict__ out,
            int DIN, int DOUT) {
  constexpr int K    = CIN * 27;
  constexpr int KP   = (K + 31) & ~31;
  constexpr int NCH  = KP / 32;
  constexpr int SPAN = 64 * MT;             // out-x covered per block
  constexpr int LX   = SPAN * STRIDE + 2;   // halo width
  constexpr int NT   = COUT / 16;
  constexpr int CV   = CIN / 8;             // 16B vectors per voxel
  __shared__ alignas(16) _Float16 sIn[9 * LX * CIN];

  const int tid  = threadIdx.x;
  const int wave = tid >> 5, lane = tid & 31;
  const int zo = blockIdx.z, yo = blockIdx.y, X0 = blockIdx.x * SPAN;

  // gfx1250 prefetch of epilogue mask row + packed weights
  __builtin_prefetch(&maskOut[(size_t)(zo * DOUT + yo) * DOUT + X0], 0, 1);
  __builtin_prefetch(wPack, 0, 1);

  // cooperative halo-tile load: NHWC rows are contiguous in (x, c) so this is
  // pure 16B vector traffic; OOB voxels are zero-filled.
  const int totalV = 9 * LX * CV;
  for (int i = tid; i < totalV; i += 128) {
    int cv = i % CV; int rest = i / CV;
    int xx = rest % LX; int r = rest / LX;
    int yyw = r % 3, zzw = r / 3;
    int zi = zo * STRIDE + zzw - 1;
    int yi = yo * STRIDE + yyw - 1;
    int xi = X0 * STRIDE + xx - 1;
    v8h v = {};
    if (zi >= 0 && zi < DIN && yi >= 0 && yi < DIN && xi >= 0 && xi < DIN)
      v = *(const v8h*)&in[(size_t)(((zi * DIN) + yi) * DIN + xi) * CIN + cv * 8];
    *(v8h*)&sIn[((zzw * 3 + yyw) * LX + xx) * CIN + cv * 8] = v;
  }
  __syncthreads();

  const int mrow = lane & 15, hl = lane >> 4;
  v8f acc[MT][NT] = {};

  auto aRun = [&](int t, int c, int xl) -> v8h {
    v8h z = {};
    if (t >= 27) return z;                       // K padding (uniform branch)
    int dz = t / 9, r9 = t % 9, dy = r9 / 3, dx = r9 % 3;
    int off = ((dz * 3 + dy) * LX + xl * STRIDE + dx) * CIN + c;
    return *(const v8h*)&sIn[off];
  };

#pragma unroll
  for (int kc = 0; kc < NCH; ++kc) {
    const int k1 = kc * 32 + hl * 8;             // first 8-K run
    const int k2 = k1 + 16;                      // second 8-K run
    const int t1 = k1 / CIN, c1 = k1 % CIN;
    const int t2 = k2 / CIN, c2 = k2 % CIN;

    v16h afr[MT];
#pragma unroll
    for (int mt = 0; mt < MT; ++mt) {
      const int xl = (wave * MT + mt) * 16 + mrow;
      afr[mt] = combine8(aRun(t1, c1, xl), aRun(t2, c2, xl));
    }
#pragma unroll
    for (int nt = 0; nt < NT; ++nt) {
      const int n = nt * 16 + mrow;
      const v8h rb1 = *(const v8h*)&wPack[n * KP + k1];
      const v8h rb2 = *(const v8h*)&wPack[n * KP + k2];
      const v16h b = combine8(rb1, rb2);
#pragma unroll
      for (int mt = 0; mt < MT; ++mt)
        acc[mt][nt] = __builtin_amdgcn_wmma_f32_16x16x32_f16(
            false, afr[mt], false, b, (short)0, acc[mt][nt], false, false);
    }
  }

  // epilogue: BN + ReLU + mask ; C/D layout: N = lane&15, M = r + 8*half.
  // NHWC store: for fixed (mt, r) the 16 lanes of a half write contiguous n.
#pragma unroll
  for (int mt = 0; mt < MT; ++mt) {
#pragma unroll
    for (int nt = 0; nt < NT; ++nt) {
      const int n = nt * 16 + mrow;
      const float s = sc[n], bv = sh[n];
#pragma unroll
      for (int r = 0; r < 8; ++r) {
        int xo = X0 + (wave * MT + mt) * 16 + r + hl * 8;
        float mv = maskOut[(size_t)(zo * DOUT + yo) * DOUT + xo];
        float v  = fmaxf(acc[mt][nt][r] * s + bv, 0.f) * mv;
        out[(size_t)((zo * DOUT + yo) * DOUT + xo) * COUT + n] = (OUT_T)v;
      }
    }
  }
}

// ---------------------------------------------------------------------------
// trilinear sampling of NHWC (64,64,64,32) volume at 65536 points -> (N, 32).
// Each corner is a 128B contiguous run -> 8x float4 loads; output is a 128B
// contiguous store per point.
__global__ void __launch_bounds__(256)
k_sample(const float* __restrict__ vol, const float* __restrict__ coords,
         float* __restrict__ out, int npts) {
  int p = blockIdx.x * blockDim.x + threadIdx.x;
  if (p >= npts) return;
  const int S = 64;
  float cx = coords[3 * p + 0], cy = coords[3 * p + 1], cz = coords[3 * p + 2];
  float fx = (cx + 1.f) * 0.5f * (S - 1);
  float fy = (cy + 1.f) * 0.5f * (S - 1);
  float fz = (cz + 1.f) * 0.5f * (S - 1);
  float x0f = floorf(fx), y0f = floorf(fy), z0f = floorf(fz);
  float tx = fx - x0f, ty = fy - y0f, tz = fz - z0f;
  int x0 = (int)x0f, y0 = (int)y0f, z0 = (int)z0f;
  v4f acc[8] = {};
#pragma unroll
  for (int c8 = 0; c8 < 8; ++c8) {
    int dx = c8 & 1, dy = (c8 >> 1) & 1, dz = c8 >> 2;
    int xi = x0 + dx, yi = y0 + dy, zi = z0 + dz;
    bool valid = (xi >= 0 && xi < S && yi >= 0 && yi < S && zi >= 0 && zi < S);
    float w = (dx ? tx : 1.f - tx) * (dy ? ty : 1.f - ty) * (dz ? tz : 1.f - tz);
    w = valid ? w : 0.f;
    int xc = min(max(xi, 0), S - 1);
    int yc = min(max(yi, 0), S - 1);
    int zc = min(max(zi, 0), S - 1);
    const v4f* s4 = (const v4f*)(vol + (size_t)((zc * S + yc) * S + xc) * 32);
#pragma unroll
    for (int q = 0; q < 8; ++q) acc[q] += w * s4[q];
  }
  v4f* o = (v4f*)(out + (size_t)p * 32);
#pragma unroll
  for (int q = 0; q < 8; ++q) o[q] = acc[q];
}

// ---------------------------------------------------------------------------
extern "C" void kernel_launch(void* const* d_in, const int* in_sizes, int n_in,
                              void* d_out, int out_size, void* d_ws, size_t ws_size,
                              hipStream_t stream) {
  (void)in_sizes; (void)n_in; (void)out_size; (void)ws_size;
  const float* x_feat = (const float*)d_in[0];
  const int*   mask   = (const int*)d_in[1];
  const float* coords = (const float*)d_in[2];
  const float* w0a = (const float*)d_in[3];
  const float* s0a = (const float*)d_in[4];
  const float* b0a = (const float*)d_in[5];
  const float* w0b = (const float*)d_in[6];
  const float* s0b = (const float*)d_in[7];
  const float* b0b = (const float*)d_in[8];
  const float* wd0 = (const float*)d_in[9];
  const float* sd0 = (const float*)d_in[10];
  const float* bd0 = (const float*)d_in[11];
  const float* w1a = (const float*)d_in[12];
  const float* s1a = (const float*)d_in[13];
  const float* b1a = (const float*)d_in[14];
  const float* w1b = (const float*)d_in[15];
  const float* s1b = (const float*)d_in[16];
  const float* b1b = (const float*)d_in[17];
  float* outp = (float*)d_out;

  const size_t V128 = 128ull * 128 * 128, V64 = 64ull * 64 * 64;
  char* p = (char*)d_ws;
  auto take = [&](size_t bytes) {
    char* r = p;
    p += (bytes + 255) & ~(size_t)255;
    return r;
  };
  float*    m0f  = (float*)take(V128 * 4);
  float*    m1f  = (float*)take(V64 * 4);
  _Float16* wpk  = (_Float16*)take(64 * 1024);     // reused packed weights
  char*     bufA = take(16 * V128 * 2);            // 67 MB ping
  char*     bufB = take(16 * V128 * 2);            // 67 MB pong
  // aliased timeline (stream-serialized): h0a=A, h0b=B, hd=A, h1a=B, h1b=A
  _Float16* h0a = (_Float16*)bufA;
  _Float16* h0b = (_Float16*)bufB;
  _Float16* hd  = (_Float16*)bufA;
  _Float16* h1a = (_Float16*)bufB;
  float*    h1b = (float*)bufA;

  const int KP16 = 448, KP32 = 864;   // (CIN*27) padded to mult. of 32

  k_mask_to_f32<<<(int)(V128 / 256), 256, 0, stream>>>(mask, m0f, (int)V128);
  k_mask_down<<<(int)(V64 / 256), 256, 0, stream>>>(m0f, m1f);
  k_conv0a<<<(int)(V128 / 256), 256, 0, stream>>>(x_feat, m0f, w0a, s0a, b0a, h0a);

  k_pack_w<<<(16 * KP16 + 255) / 256, 256, 0, stream>>>(w0b, wpk, 16, 16, KP16);
  k_conv_wmma<16, 16, 1, 2, _Float16>
      <<<dim3(1, 128, 128), 128, 0, stream>>>(h0a, wpk, s0b, b0b, m0f, h0b, 128, 128);

  k_pack_w<<<(32 * KP16 + 255) / 256, 256, 0, stream>>>(wd0, wpk, 16, 32, KP16);
  k_conv_wmma<16, 32, 2, 1, _Float16>
      <<<dim3(1, 64, 64), 128, 0, stream>>>(h0b, wpk, sd0, bd0, m1f, hd, 128, 64);

  k_pack_w<<<(32 * KP32 + 255) / 256, 256, 0, stream>>>(w1a, wpk, 32, 32, KP32);
  k_conv_wmma<32, 32, 1, 1, _Float16>
      <<<dim3(1, 64, 64), 128, 0, stream>>>(hd, wpk, s1a, b1a, m1f, h1a, 64, 64);

  k_pack_w<<<(32 * KP32 + 255) / 256, 256, 0, stream>>>(w1b, wpk, 32, 32, KP32);
  k_conv_wmma<32, 32, 1, 1, float>
      <<<dim3(1, 64, 64), 128, 0, stream>>>(h1a, wpk, s1b, b1b, m1f, h1b, 64, 64);

  k_sample<<<65536 / 256, 256, 0, stream>>>(h1b, coords, outp, 65536);
}
